// MambaLayer_6373731467393
// MI455X (gfx1250) — matrix operand order repaired and checked by reading, not compiled, not run on hardware
//
#include <hip/hip_runtime.h>
#include <hip/hip_bf16.h>
#include <math.h>

#define DIMC     128
#define DSTATE   16
#define DCONV    4
#define DINNER   256
#define P2       512      // 2*D_INNER
#define DTRANK   8
#define NSL      16
#define BB       4
#define LL       4096
#define MSL      (LL / NSL)   // 256

typedef __attribute__((ext_vector_type(16))) _Float16 v16h;
typedef __attribute__((ext_vector_type(8)))  float    v8f;

// scan-order position l -> original xz position, per direction
__device__ __forceinline__ int map_idx(int dir, int l) {
    if (dir == 0) return l;
    if (dir == 1) return LL - 1 - l;
    int s = l & (NSL - 1);   // l % 16
    int j = l >> 4;          // l / 16
    return s * MSL + j;
}

// -------------------------------------------------------------------------
// WMMA GEMM, K-contiguous operands:
//   C[m][n] = sum_k A[m*lda + k] * B[n*ldb + k]      (B is N-major, K-contig)
// One wave computes a 16(M) x 64(N) strip: A fragment loaded once per K-step
// (4x float4), reused across 4 accumulators / 4 v_wmma_f32_16x16x32_f16.
// B fragments are 4x float4 each. Fragment layouts per CDNA5 ISA §7.12.2.
// -------------------------------------------------------------------------
__global__ void wmma_gemm_kernel(const float* __restrict__ A, long aBatch, int lda,
                                 const float* __restrict__ Bp, long bBatch, int ldb,
                                 float* __restrict__ Cp, long cBatch, int ldc,
                                 int M, int N, int K)
{
    const int lane   = threadIdx.x;          // 0..31, one wave per block
    const int strips = N >> 6;               // 64-wide N strips
    const int tm     = (blockIdx.x / strips) << 4;
    const int tn0    = (blockIdx.x % strips) << 6;
    const float* Ab  = A  + (long)blockIdx.y * aBatch;
    const float* Bb  = Bp + (long)blockIdx.y * bBatch;
    float*       Cb  = Cp + (long)blockIdx.y * cBatch;

    const int half = lane >> 4;              // 0: lanes 0-15, 1: lanes 16-31
    const int l16  = lane & 15;

    const int arow = tm + l16;               // A: row M = lane&15
    const int akb  = half ? 8  : 0;          // A K sub-base per 16-group
    const int bkb  = half ? 8  : 0;          // B K base

    v8f acc0 = {}, acc1 = {}, acc2 = {}, acc3 = {};

    for (int k0 = 0; k0 < K; k0 += 32) {
        // ---- A fragment: rows contiguous in K -> 4x b128 loads ----
        const float* arP = Ab + (long)arow * lda + k0;
        float4 a0 = *(const float4*)(arP + akb);
        float4 a1 = *(const float4*)(arP + akb + 4);
        float4 a2 = *(const float4*)(arP + 16 + akb);
        float4 a3 = *(const float4*)(arP + 16 + akb + 4);
        v16h a;
        a[0]=(_Float16)a0.x; a[1]=(_Float16)a0.y; a[2]=(_Float16)a0.z; a[3]=(_Float16)a0.w;
        a[4]=(_Float16)a1.x; a[5]=(_Float16)a1.y; a[6]=(_Float16)a1.z; a[7]=(_Float16)a1.w;
        a[8]=(_Float16)a2.x; a[9]=(_Float16)a2.y; a[10]=(_Float16)a2.z; a[11]=(_Float16)a2.w;
        a[12]=(_Float16)a3.x; a[13]=(_Float16)a3.y; a[14]=(_Float16)a3.z; a[15]=(_Float16)a3.w;

        // ---- B fragments: column (n) fixed, K contiguous -> 4x b128 each ----
#pragma unroll
        for (int t = 0; t < 4; ++t) {
            const float* bcP = Bb + (long)(tn0 + (t << 4) + l16) * ldb + k0 + bkb;
            float4 b0 = *(const float4*)(bcP);
            float4 b1 = *(const float4*)(bcP + 4);
            float4 b2 = *(const float4*)(bcP + 16);
            float4 b3 = *(const float4*)(bcP + 20);
            v16h bf;
            bf[0]=(_Float16)b0.x; bf[1]=(_Float16)b0.y; bf[2]=(_Float16)b0.z; bf[3]=(_Float16)b0.w;
            bf[4]=(_Float16)b1.x; bf[5]=(_Float16)b1.y; bf[6]=(_Float16)b1.z; bf[7]=(_Float16)b1.w;
            bf[8]=(_Float16)b2.x; bf[9]=(_Float16)b2.y; bf[10]=(_Float16)b2.z; bf[11]=(_Float16)b2.w;
            bf[12]=(_Float16)b3.x; bf[13]=(_Float16)b3.y; bf[14]=(_Float16)b3.z; bf[15]=(_Float16)b3.w;
            if      (t == 0) acc0 = __builtin_amdgcn_wmma_f32_16x16x32_f16(false, a, false, bf, (short)0, acc0, false, false);
            else if (t == 1) acc1 = __builtin_amdgcn_wmma_f32_16x16x32_f16(false, a, false, bf, (short)0, acc1, false, false);
            else if (t == 2) acc2 = __builtin_amdgcn_wmma_f32_16x16x32_f16(false, a, false, bf, (short)0, acc2, false, false);
            else             acc3 = __builtin_amdgcn_wmma_f32_16x16x32_f16(false, a, false, bf, (short)0, acc3, false, false);
        }
    }

#pragma unroll
    for (int j = 0; j < 8; ++j) {
        const long mrow = (long)(tm + j + (half << 3)) * ldc;
        Cb[mrow + tn0 +  0 + l16] = acc0[j];
        Cb[mrow + tn0 + 16 + l16] = acc1[j];
        Cb[mrow + tn0 + 32 + l16] = acc2[j];
        Cb[mrow + tn0 + 48 + l16] = acc3[j];
    }
}

// -------------------------------------------------------------------------
// LDS-tiled transpose: (B, DINNER, L) -> (B, L, DINNER)
// -------------------------------------------------------------------------
__global__ void transpose_kernel(const float* __restrict__ src,
                                 float* __restrict__ dst)
{
    __shared__ float tile[16][17];
    const int b  = blockIdx.z;
    const int d0 = blockIdx.x << 4;   // over DINNER
    const int l0 = blockIdx.y << 4;   // over L
    const int tx = threadIdx.x, ty = threadIdx.y;
    tile[ty][tx] = src[((long)b * DINNER + d0 + ty) * LL + l0 + tx];
    __syncthreads();
    dst[((long)b * LL + l0 + ty) * DINNER + d0 + tx] = tile[tx][ty];
}

// -------------------------------------------------------------------------
// Causal conv1d (scan order) + SiLU, and gate (z) extraction.
// -------------------------------------------------------------------------
__global__ void prep_kernel(const float* __restrict__ xz,
                            const float* __restrict__ conv_w,  // (256,4)
                            const float* __restrict__ conv_b,  // (256)
                            float* __restrict__ xo,            // (B,256,L)
                            float* __restrict__ zo,            // (B,256,L)
                            int dir)
{
    const int b = blockIdx.x / LL;
    const int l = blockIdx.x % LL;
    const int d = threadIdx.x;
    float acc = conv_b[d];
#pragma unroll
    for (int k = 0; k < DCONV; ++k) {
        int ls = l - (DCONV - 1) + k;
        if (ls >= 0)
            acc += conv_w[d * DCONV + k] *
                   xz[((long)b * P2 + d) * LL + map_idx(dir, ls)];
    }
    float xs = acc / (1.f + __expf(-acc));             // SiLU
    long  o  = ((long)b * DINNER + d) * LL + l;
    xo[o] = xs;
    zo[o] = xz[((long)b * P2 + DINNER + d) * LL + map_idx(dir, l)];
}

// -------------------------------------------------------------------------
// x_proj: x_dbl[b,r,l] = sum_d x_proj_w[r,d] * x[b,d,l]; split dt/B/C.
// -------------------------------------------------------------------------
__global__ void xproj_kernel(const float* __restrict__ xpw,   // (40,256)
                             const float* __restrict__ x,
                             float* __restrict__ dt_raw,      // (B,8,L)
                             float* __restrict__ Bm,          // (B,16,L)
                             float* __restrict__ Cm)          // (B,16,L)
{
    const int b = blockIdx.x / LL;
    const int l = blockIdx.x % LL;
    const int r = threadIdx.x;
    if (r >= DTRANK + 2 * DSTATE) return;
    float acc = 0.f;
    for (int d = 0; d < DINNER; ++d)
        acc += xpw[r * DINNER + d] * x[((long)b * DINNER + d) * LL + l];
    if (r < DTRANK)
        dt_raw[((long)b * DTRANK + r) * LL + l] = acc;
    else if (r < DTRANK + DSTATE)
        Bm[((long)b * DSTATE + (r - DTRANK)) * LL + l] = acc;
    else
        Cm[((long)b * DSTATE + (r - DTRANK - DSTATE)) * LL + l] = acc;
}

// -------------------------------------------------------------------------
// delta[b,d,l] = softplus(dt_w[d,:] . dt_raw[b,:,l] + dt_b[d])
// -------------------------------------------------------------------------
__global__ void delta_kernel(const float* __restrict__ dtw,   // (256,8)
                             const float* __restrict__ dtb,   // (256)
                             const float* __restrict__ dt_raw,
                             float* __restrict__ delta)
{
    const int b = blockIdx.x / LL;
    const int l = blockIdx.x % LL;
    const int d = threadIdx.x;
    float acc = dtb[d];
#pragma unroll
    for (int r = 0; r < DTRANK; ++r)
        acc += dtw[d * DTRANK + r] * dt_raw[((long)b * DTRANK + r) * LL + l];
    float sp = (acc > 20.f) ? acc : __logf(1.f + __expf(acc));
    delta[((long)b * DINNER + d) * LL + l] = sp;
}

// -------------------------------------------------------------------------
// Selective scan: one thread per (d,n) state cell; recurrence in registers;
// 16-lane shuffle reduction over n; accumulate (with D skip + SiLU gate)
// into y_tot at the inverse-mapped original position.
// -------------------------------------------------------------------------
__global__ void scan_kernel(const float* __restrict__ u,
                            const float* __restrict__ delta,
                            const float* __restrict__ A_log,  // (256,16)
                            const float* __restrict__ Bm,
                            const float* __restrict__ Cm,
                            const float* __restrict__ Dp,     // (256)
                            const float* __restrict__ z,
                            float* __restrict__ y_acc,
                            int dir)
{
    const int b     = blockIdx.x >> 4;
    const int dtile = blockIdx.x & 15;
    const int n     = threadIdx.x & 15;
    const int d     = (dtile << 4) + (threadIdx.x >> 4);

    const float A  = -__expf(A_log[d * DSTATE + n]);
    const float Dd = Dp[d];
    const float* uP  = u     + ((long)b * DINNER + d) * LL;
    const float* deP = delta + ((long)b * DINNER + d) * LL;
    const float* bP  = Bm    + ((long)b * DSTATE + n) * LL;
    const float* cP  = Cm    + ((long)b * DSTATE + n) * LL;
    const float* zP  = z     + ((long)b * DINNER + d) * LL;

    float h = 0.f;
    for (int l = 0; l < LL; ++l) {
        float de = deP[l];
        float uu = uP[l];
        float hb = bP[l];
        float cc = cP[l];
        __builtin_prefetch(&deP[l + 16], 0, 0);
        __builtin_prefetch(&uP[l + 16], 0, 0);

        h = h * __expf(de * A) + de * uu * hb;
        float yp = h * cc;
#pragma unroll
        for (int off = 8; off > 0; off >>= 1)
            yp += __shfl_xor(yp, off, 16);
        if (n == 0) {
            float y  = yp + Dd * uu;
            float zz = zP[l];
            float g  = zz / (1.f + __expf(-zz));
            y_acc[((long)b * DINNER + d) * LL + map_idx(dir, l)] += y * g;
        }
    }
}

__global__ void zero_kernel(float* __restrict__ p, long nElem)
{
    long i = (long)blockIdx.x * blockDim.x + threadIdx.x;
    if (i < nElem) p[i] = 0.f;
}

// -------------------------------------------------------------------------
extern "C" void kernel_launch(void* const* d_in, const int* in_sizes, int n_in,
                              void* d_out, int out_size, void* d_ws, size_t ws_size,
                              hipStream_t stream)
{
    (void)in_sizes; (void)n_in; (void)out_size; (void)ws_size;

    const float* x_in       = (const float*)d_in[0];
    const float* in_proj_w  = (const float*)d_in[1];
    const float* conv_w[3]  = {(const float*)d_in[2],  (const float*)d_in[9],  (const float*)d_in[16]};
    const float* conv_b[3]  = {(const float*)d_in[3],  (const float*)d_in[10], (const float*)d_in[17]};
    const float* x_proj[3]  = {(const float*)d_in[4],  (const float*)d_in[11], (const float*)d_in[18]};
    const float* dt_w[3]    = {(const float*)d_in[5],  (const float*)d_in[12], (const float*)d_in[19]};
    const float* dt_b[3]    = {(const float*)d_in[6],  (const float*)d_in[13], (const float*)d_in[20]};
    const float* A_log[3]   = {(const float*)d_in[7],  (const float*)d_in[14], (const float*)d_in[21]};
    const float* Dp[3]      = {(const float*)d_in[8],  (const float*)d_in[15], (const float*)d_in[22]};
    const float* out_proj_w = (const float*)d_in[23];
    float* out = (float*)d_out;

    // workspace carve-up (floats)
    float* ws     = (float*)d_ws;
    long   off    = 0;
    float* xz     = ws + off; off += (long)BB * P2 * LL;       // 32 MB
    float* xbuf   = ws + off; off += (long)BB * DINNER * LL;   // 16 MB
    float* zbuf   = ws + off; off += (long)BB * DINNER * LL;   // 16 MB
    float* dt_raw = ws + off; off += (long)BB * DTRANK * LL;
    float* Bm     = ws + off; off += (long)BB * DSTATE * LL;
    float* Cm     = ws + off; off += (long)BB * DSTATE * LL;
    float* delta  = ws + off; off += (long)BB * DINNER * LL;   // 16 MB
    float* y_tot  = ws + off; off += (long)BB * DINNER * LL;   // 16 MB
    float* y_t    = ws + off; off += (long)BB * DINNER * LL;   // 16 MB transposed

    // in_proj: xz[b,p,l] = sum_d W[p,d] * x_in[b,l,d]
    //   A = W (512x128, no batch), B[k][n] = x_in[b,n,k] (K contiguous)
    {
        dim3 grid((P2 / 16) * (LL / 64), BB);
        wmma_gemm_kernel<<<grid, 32, 0, stream>>>(
            in_proj_w, 0, DIMC,
            x_in, (long)LL * DIMC, DIMC,
            xz,   (long)P2 * LL,   LL,
            P2, LL, DIMC);
    }

    const long ytotN = (long)BB * DINNER * LL;
    zero_kernel<<<(unsigned)((ytotN + 255) / 256), 256, 0, stream>>>(y_tot, ytotN);

    for (int dir = 0; dir < 3; ++dir) {
        prep_kernel<<<BB * LL, DINNER, 0, stream>>>(
            xz, conv_w[dir], conv_b[dir], xbuf, zbuf, dir);
        xproj_kernel<<<BB * LL, 64, 0, stream>>>(
            x_proj[dir], xbuf, dt_raw, Bm, Cm);
        delta_kernel<<<BB * LL, DINNER, 0, stream>>>(
            dt_w[dir], dt_b[dir], dt_raw, delta);
        scan_kernel<<<BB * 16, 256, 0, stream>>>(
            xbuf, delta, A_log[dir], Bm, Cm, Dp[dir], zbuf, y_tot, dir);
    }

    // transpose y_tot (B,256,L) -> y_t (B,L,256) so out_proj is K-contiguous
    {
        dim3 grid(DINNER / 16, LL / 16, BB);
        dim3 blk(16, 16);
        transpose_kernel<<<grid, blk, 0, stream>>>(y_tot, y_t);
    }

    // out_proj: out[b,l,o] = sum_d y_t[b,l,d] * W[o,d]
    //   A = y_t (4096x256 per batch), B[k][n] = W[n*256+k] (K contiguous, no batch)
    {
        dim3 grid((LL / 16) * (DIMC / 64), BB);
        wmma_gemm_kernel<<<grid, 32, 0, stream>>>(
            y_t, (long)LL * DINNER, DINNER,
            out_proj_w, 0, DINNER,
            out, (long)LL * DIMC, DIMC,
            LL, DIMC, DINNER);
    }
}
